// Attention_28140625723842
// MI455X (gfx1250) — matrix operand order repaired and checked by reading, not compiled, not run on hardware
//
#include <hip/hip_runtime.h>
#include <hip/hip_bf16.h>
#include <stdint.h>

#define BB 2
#define SS 2048
#define DD 1024
#define HH 16
#define DHH 64

typedef __attribute__((ext_vector_type(16))) __bf16 v16bf;
typedef __attribute__((ext_vector_type(8)))  __bf16 v8bf;
typedef __attribute__((ext_vector_type(8)))  float  v8f;
typedef __attribute__((ext_vector_type(4)))  float  v4f;

static __device__ __forceinline__ v16bf join8(v8bf lo, v8bf hi) {
  return __builtin_shufflevector(lo, hi, 0,1,2,3,4,5,6,7,8,9,10,11,12,13,14,15);
}
static __device__ __forceinline__ v8f vzero8() {
  v8f z = {0.f,0.f,0.f,0.f,0.f,0.f,0.f,0.f};
  return z;
}

// ---------------------------------------------------------------------------
// Kernel 1: transpose + convert Wq/Wk/Wv (f32 [k][n]) -> bf16 [m][n][k]
// so WMMA B-tiles (one column per lane, two runs of 8 consecutive K) are two
// 16-byte loads.
// ---------------------------------------------------------------------------
__global__ void __launch_bounds__(256) prep_weights_kernel(
    const float* __restrict__ Wq, const float* __restrict__ Wk,
    const float* __restrict__ Wv, __bf16* __restrict__ wt) {
  int idx = blockIdx.x * blockDim.x + threadIdx.x;   // 3 * 1024 * 1024 threads
  int m   = idx >> 20;
  int rem = idx & ((1 << 20) - 1);
  int k   = rem >> 10;
  int n   = rem & 1023;
  const float* W = (m == 0) ? Wq : ((m == 1) ? Wk : Wv);
  wt[(size_t)m * DD * DD + (size_t)n * DD + k] = (__bf16)W[(size_t)k * DD + n];
}

// ---------------------------------------------------------------------------
// Kernel 2: projections X @ W + b, one wave per 16x64 tile. K-loop unrolled
// x2 with ping-pong register buffer sets: each set is reloaded in place right
// after consumption, so no register rotation moves and every WMMA consumes
// loads issued a full 64-deep K-chunk earlier.
// Outputs: Qp,Kp bf16 [b,h,s,dh]; Vt bf16 [b,h,dh,s] (transposed for P@V).
// ---------------------------------------------------------------------------
__global__ void __launch_bounds__(256) proj_kernel(
    const float* __restrict__ q, const float* __restrict__ k,
    const float* __restrict__ v, const __bf16* __restrict__ wt,
    const float* __restrict__ bq, const float* __restrict__ bk,
    const float* __restrict__ bv,
    __bf16* __restrict__ Qp, __bf16* __restrict__ Kp, __bf16* __restrict__ Vt) {
  const int lane = threadIdx.x & 31;
  const int wave = (blockIdx.x * blockDim.x + threadIdx.x) >> 5;
  const int m    = wave >> 12;          // which projection (0=q,1=k,2=v)
  const int t    = wave & 4095;
  const int mt   = t >> 4;              // row tile 0..255
  const int nt   = t & 15;              // col tile 0..15 (64 cols)
  const int l16  = lane & 15;
  const int half = lane >> 4;
  const int nbase = nt * 64;
  const int row   = mt * 16 + l16;      // A-layout: M = lane % 16

  const float*  X    = (m == 0) ? q  : ((m == 1) ? k  : v);
  const float*  bias = (m == 0) ? bq : ((m == 1) ? bk : bv);
  const __bf16* W    = wt + (size_t)m * DD * DD;
  const float*  xrow = X + (size_t)row * DD;

  auto loadA = [&](int k0) -> v16bf {
    v4f f0 = *(const v4f*)(xrow + k0 + half * 8);
    v4f f1 = *(const v4f*)(xrow + k0 + half * 8 + 4);
    v4f f2 = *(const v4f*)(xrow + k0 + 16 + half * 8);
    v4f f3 = *(const v4f*)(xrow + k0 + 16 + half * 8 + 4);
    v16bf a;
    #pragma unroll
    for (int e = 0; e < 4; ++e) {
      a[e]      = (__bf16)f0[e];
      a[e + 4]  = (__bf16)f1[e];
      a[e + 8]  = (__bf16)f2[e];
      a[e + 12] = (__bf16)f3[e];
    }
    return a;
  };
  auto loadB = [&](int k0, int nc) -> v16bf {
    const __bf16* wrow =
        W + (size_t)(nbase + nc * 16 + l16) * DD + k0 + half * 8;
    return join8(*(const v8bf*)(wrow), *(const v8bf*)(wrow + 16));
  };

  v8f acc[4];
  #pragma unroll
  for (int nc = 0; nc < 4; ++nc) acc[nc] = vzero8();

  // Ping-pong buffer sets A (even chunks) and B (odd chunks).
  v16bf aA = loadA(0);
  v16bf bA0 = loadB(0, 0), bA1 = loadB(0, 1), bA2 = loadB(0, 2),
        bA3 = loadB(0, 3);
  v16bf aB = loadA(32);
  v16bf bB0 = loadB(32, 0), bB1 = loadB(32, 1), bB2 = loadB(32, 2),
        bB3 = loadB(32, 3);

  for (int k0 = 0; k0 < DD - 64; k0 += 64) {
    acc[0] = __builtin_amdgcn_wmma_f32_16x16x32_bf16(false, aA, false, bA0,
                                                     (short)0, acc[0], false, false);
    acc[1] = __builtin_amdgcn_wmma_f32_16x16x32_bf16(false, aA, false, bA1,
                                                     (short)0, acc[1], false, false);
    acc[2] = __builtin_amdgcn_wmma_f32_16x16x32_bf16(false, aA, false, bA2,
                                                     (short)0, acc[2], false, false);
    acc[3] = __builtin_amdgcn_wmma_f32_16x16x32_bf16(false, aA, false, bA3,
                                                     (short)0, acc[3], false, false);
    aA  = loadA(k0 + 64);
    bA0 = loadB(k0 + 64, 0); bA1 = loadB(k0 + 64, 1);
    bA2 = loadB(k0 + 64, 2); bA3 = loadB(k0 + 64, 3);

    acc[0] = __builtin_amdgcn_wmma_f32_16x16x32_bf16(false, aB, false, bB0,
                                                     (short)0, acc[0], false, false);
    acc[1] = __builtin_amdgcn_wmma_f32_16x16x32_bf16(false, aB, false, bB1,
                                                     (short)0, acc[1], false, false);
    acc[2] = __builtin_amdgcn_wmma_f32_16x16x32_bf16(false, aB, false, bB2,
                                                     (short)0, acc[2], false, false);
    acc[3] = __builtin_amdgcn_wmma_f32_16x16x32_bf16(false, aB, false, bB3,
                                                     (short)0, acc[3], false, false);
    aB  = loadA(k0 + 96);
    bB0 = loadB(k0 + 96, 0); bB1 = loadB(k0 + 96, 1);
    bB2 = loadB(k0 + 96, 2); bB3 = loadB(k0 + 96, 3);
  }
  // Tail: consume the last two preloaded chunks.
  acc[0] = __builtin_amdgcn_wmma_f32_16x16x32_bf16(false, aA, false, bA0,
                                                   (short)0, acc[0], false, false);
  acc[1] = __builtin_amdgcn_wmma_f32_16x16x32_bf16(false, aA, false, bA1,
                                                   (short)0, acc[1], false, false);
  acc[2] = __builtin_amdgcn_wmma_f32_16x16x32_bf16(false, aA, false, bA2,
                                                   (short)0, acc[2], false, false);
  acc[3] = __builtin_amdgcn_wmma_f32_16x16x32_bf16(false, aA, false, bA3,
                                                   (short)0, acc[3], false, false);
  acc[0] = __builtin_amdgcn_wmma_f32_16x16x32_bf16(false, aB, false, bB0,
                                                   (short)0, acc[0], false, false);
  acc[1] = __builtin_amdgcn_wmma_f32_16x16x32_bf16(false, aB, false, bB1,
                                                   (short)0, acc[1], false, false);
  acc[2] = __builtin_amdgcn_wmma_f32_16x16x32_bf16(false, aB, false, bB2,
                                                   (short)0, acc[2], false, false);
  acc[3] = __builtin_amdgcn_wmma_f32_16x16x32_bf16(false, aB, false, bB3,
                                                   (short)0, acc[3], false, false);

  // Epilogue: bias add, convert to bf16, scatter into head-split layouts.
  #pragma unroll
  for (int nc = 0; nc < 4; ++nc) {
    int   n   = nbase + nc * 16 + l16;
    float bv_ = bias[n];
    int   h   = n >> 6;
    int   dh  = n & 63;
    #pragma unroll
    for (int r = 0; r < 8; ++r) {
      int    grow = mt * 16 + r + 8 * half;   // C-layout: M = r + 8*(lane/16)
      int    bidx = grow >> 11;
      int    s    = grow & 2047;
      float  val  = acc[nc][r] + bv_;
      size_t bh   = (size_t)bidx * HH + h;
      if (m == 0)      Qp[(bh * SS + s) * DHH + dh] = (__bf16)val;
      else if (m == 1) Kp[(bh * SS + s) * DHH + dh] = (__bf16)val;
      else             Vt[(bh * DHH + dh) * SS + s] = (__bf16)val;
    }
  }
}

// ---------------------------------------------------------------------------
// Kernel 3: flash-attention, one wave per (b,h,16-row Q tile).
// K/V chunks are streamed through LDS double buffers with CDNA5 async copies
// (global_load_async_to_lds_b128 + s_wait_asynccnt) so the WMMA pipe consumes
// chunk i from LDS while chunk i+1 is in flight.
//   ldsK rows padded to 144B, ldsV rows to 80B: 16B-aligned ds_load_b128
//   B-tile reads hit 16 distinct bank groups (conflict-free).
// ---------------------------------------------------------------------------
__global__ void __launch_bounds__(64) attn_kernel(
    const __bf16* __restrict__ Qp, const __bf16* __restrict__ Kp,
    const __bf16* __restrict__ Vt, const int* __restrict__ mask,
    float* __restrict__ out) {
  __shared__ __bf16 ldsK[2][2][32 * 72];   // [wave][buf] 32 key rows x 144B
  __shared__ __bf16 ldsV[2][2][64 * 40];   // [wave][buf] 64 d rows   x  80B
  __shared__ __bf16 ldsP[2][16 * 40];      // [wave] P staging, 80B row stride

  const int lane = threadIdx.x & 31;
  const int wv   = threadIdx.x >> 5;
  const int wave = (blockIdx.x * blockDim.x + threadIdx.x) >> 5;  // 0..4095
  const int bh   = wave >> 7;                                     // 0..31
  const int qt   = wave & 127;
  const int qbase = qt * 16;
  const int b    = bh >> 4;
  const int h    = bh & 15;
  const int l16  = lane & 15;
  const int half = lane >> 4;
  const float scale = 0.03125f;            // 1/sqrt(D) = 1/32

  // Q tile resident in registers (A-layout, DH=64 -> two 16x32 tiles).
  const __bf16* qrow = Qp + ((size_t)bh * SS + qbase + l16) * DHH;
  v16bf a0 = join8(*(const v8bf*)(qrow + half * 16),
                   *(const v8bf*)(qrow + half * 16 + 8));
  v16bf a1 = join8(*(const v8bf*)(qrow + 32 + half * 16),
                   *(const v8bf*)(qrow + 32 + half * 16 + 8));

  v8f o[4];
  #pragma unroll
  for (int ch = 0; ch < 4; ++ch) o[ch] = vzero8();
  float mrow[8], lrow[8];
  #pragma unroll
  for (int r = 0; r < 8; ++r) { mrow[r] = -1e30f; lrow[r] = 0.0f; }

  const uint64_t kgb = (uint64_t)(uintptr_t)(Kp + (size_t)bh * SS * DHH);
  const uint64_t vgb = (uint64_t)(uintptr_t)(Vt + (size_t)bh * DHH * SS);
  const int*     mkp = mask + (size_t)b * SS;

  // --- async chunk issue: 8 + 8 x b128 per 32-key chunk (ASYNCcnt +16) ----
  auto issueK = [&](int kb, int buf) {
    const int r4 = lane >> 3, c = lane & 7;   // 4 rows/op, 8 lanes/row
    uint32_t lb = (uint32_t)(uintptr_t)&ldsK[wv][buf][0];
    uint64_t gb = kgb + (uint64_t)kb * (DHH * 2);
    #pragma unroll
    for (int i = 0; i < 8; ++i) {
      uint32_t lo = lb + (uint32_t)((i * 4 + r4) * 144 + c * 16);
      uint64_t go = gb + (uint64_t)((i * 4 + r4) * 128 + c * 16);
      asm volatile("global_load_async_to_lds_b128 %0, %1, off"
                   :: "v"(lo), "v"(go) : "memory");
    }
  };
  auto issueV = [&](int kb, int buf) {
    const int r8 = lane >> 2, c = lane & 3;   // 8 d-rows/op, 4 lanes/row
    uint32_t lb = (uint32_t)(uintptr_t)&ldsV[wv][buf][0];
    uint64_t gb = vgb + (uint64_t)kb * 2;
    #pragma unroll
    for (int i = 0; i < 8; ++i) {
      uint32_t lo = lb + (uint32_t)((i * 8 + r8) * 80 + c * 16);
      uint64_t go = gb + (uint64_t)(i * 8 + r8) * (SS * 2) + (uint64_t)(c * 16);
      asm volatile("global_load_async_to_lds_b128 %0, %1, off"
                   :: "v"(lo), "v"(go) : "memory");
    }
  };

  issueK(0, 0);
  issueV(0, 0);

  for (int kb = 0; kb < SS; kb += 32) {
    const int buf = (kb >> 5) & 1;
    if (kb + 32 < SS) {
      issueK(kb + 32, buf ^ 1);
      issueV(kb + 32, buf ^ 1);
      // 16 newer ops may stay in flight; current chunk's 16 are complete.
      asm volatile("s_wait_asynccnt 0x10" ::: "memory");
    } else {
      asm volatile("s_wait_asynccnt 0x0" ::: "memory");
    }

    const __bf16* kl = &ldsK[wv][buf][0];
    const __bf16* vl = &ldsV[wv][buf][0];

    // Scores: two 16x16 tiles (keys +0..15, +16..31); K-dim 64 = 2 WMMA each.
    v16bf bk00 = join8(*(const v8bf*)(kl + l16 * 72 + half * 16),
                       *(const v8bf*)(kl + l16 * 72 + half * 16 + 8));
    v16bf bk01 = join8(*(const v8bf*)(kl + l16 * 72 + 32 + half * 16),
                       *(const v8bf*)(kl + l16 * 72 + 32 + half * 16 + 8));
    v16bf bk10 = join8(*(const v8bf*)(kl + (16 + l16) * 72 + half * 16),
                       *(const v8bf*)(kl + (16 + l16) * 72 + half * 16 + 8));
    v16bf bk11 = join8(*(const v8bf*)(kl + (16 + l16) * 72 + 32 + half * 16),
                       *(const v8bf*)(kl + (16 + l16) * 72 + 32 + half * 16 + 8));

    v8f s0v = vzero8(), s1v = vzero8();
    s0v = __builtin_amdgcn_wmma_f32_16x16x32_bf16(false, a0, false, bk00,
                                                  (short)0, s0v, false, false);
    s0v = __builtin_amdgcn_wmma_f32_16x16x32_bf16(false, a1, false, bk01,
                                                  (short)0, s0v, false, false);
    s1v = __builtin_amdgcn_wmma_f32_16x16x32_bf16(false, a0, false, bk10,
                                                  (short)0, s1v, false, false);
    s1v = __builtin_amdgcn_wmma_f32_16x16x32_bf16(false, a1, false, bk11,
                                                  (short)0, s1v, false, false);

    float bias0 = mkp[kb + l16]      ? 0.0f : -1e9f;
    float bias1 = mkp[kb + 16 + l16] ? 0.0f : -1e9f;

    // Online softmax. C-layout row M = r + 8*half lives across one 16-lane
    // half; xor-shuffles 1,2,4,8 reduce over key columns.
    float p0[8], p1[8];
    #pragma unroll
    for (int r = 0; r < 8; ++r) {
      float x0 = s0v[r] * scale + bias0;
      float x1 = s1v[r] * scale + bias1;
      float t = fmaxf(x0, x1);
      t = fmaxf(t, __shfl_xor(t, 1, 32));
      t = fmaxf(t, __shfl_xor(t, 2, 32));
      t = fmaxf(t, __shfl_xor(t, 4, 32));
      t = fmaxf(t, __shfl_xor(t, 8, 32));
      float mn = fmaxf(mrow[r], t);
      float al = __expf(mrow[r] - mn);
      mrow[r] = mn;
      float e0 = __expf(x0 - mn);
      float e1 = __expf(x1 - mn);
      float rs = e0 + e1;
      rs += __shfl_xor(rs, 1, 32);
      rs += __shfl_xor(rs, 2, 32);
      rs += __shfl_xor(rs, 4, 32);
      rs += __shfl_xor(rs, 8, 32);
      lrow[r] = lrow[r] * al + rs;
      p0[r] = e0;
      p1[r] = e1;
      o[0][r] *= al; o[1][r] *= al; o[2][r] *= al; o[3][r] *= al;
    }

    // C-layout -> 16-bit A-layout rearrange via per-wave LDS tile.
    __bf16* pl = &ldsP[wv][0];
    #pragma unroll
    for (int r = 0; r < 8; ++r) {
      int rr = r + 8 * half;
      pl[rr * 40 + l16]      = (__bf16)p0[r];
      pl[rr * 40 + 16 + l16] = (__bf16)p1[r];
    }
    asm volatile("s_wait_dscnt 0x0" ::: "memory");  // DS RAW fence (same wave)
    const __bf16* pr = pl + l16 * 40;
    v16bf pa = join8(*(const v8bf*)(pr + half * 16),
                     *(const v8bf*)(pr + half * 16 + 8));

    // O += P @ V from the LDS V chunk (B-layout reads, conflict-free).
    #pragma unroll
    for (int ch = 0; ch < 4; ++ch) {
      v16bf bv16 = join8(*(const v8bf*)(vl + (ch * 16 + l16) * 40 + half * 16),
                         *(const v8bf*)(vl + (ch * 16 + l16) * 40 + half * 16 + 8));
      o[ch] = __builtin_amdgcn_wmma_f32_16x16x32_bf16(
          false, pa, false, bv16, (short)0, o[ch], false, false);
    }
  }

  // Normalize (one reciprocal per row) and write f32 [B,S,D] output.
  float inv[8];
  #pragma unroll
  for (int r = 0; r < 8; ++r) inv[r] = 1.0f / lrow[r];
  #pragma unroll
  for (int ch = 0; ch < 4; ++ch) {
    int col = h * 64 + ch * 16 + l16;
    #pragma unroll
    for (int r = 0; r < 8; ++r) {
      int s = qbase + r + 8 * half;
      out[((size_t)b * SS + s) * DD + col] = o[ch][r] * inv[r];
    }
  }
}

// ---------------------------------------------------------------------------
extern "C" void kernel_launch(void* const* d_in, const int* in_sizes, int n_in,
                              void* d_out, int out_size, void* d_ws,
                              size_t ws_size, hipStream_t stream) {
  const float* q    = (const float*)d_in[0];
  const float* k    = (const float*)d_in[1];
  const float* v    = (const float*)d_in[2];
  const int*   mask = (const int*)  d_in[3];
  const float* Wq   = (const float*)d_in[4];
  const float* bq   = (const float*)d_in[5];
  const float* Wk   = (const float*)d_in[6];
  const float* bk   = (const float*)d_in[7];
  const float* Wv   = (const float*)d_in[8];
  const float* bv   = (const float*)d_in[9];
  float*       out  = (float*)d_out;

  char* ws = (char*)d_ws;
  const size_t wt_bytes   = (size_t)3 * DD * DD * 2;   // bf16 weights (6 MB)
  const size_t proj_bytes = (size_t)BB * SS * DD * 2;  // 8 MB each
  __bf16* wt = (__bf16*)ws;
  __bf16* Qp = (__bf16*)(ws + wt_bytes);
  __bf16* Kp = (__bf16*)(ws + wt_bytes + proj_bytes);
  __bf16* Vt = (__bf16*)(ws + wt_bytes + 2 * proj_bytes);

  // 1) weight transpose+convert
  prep_weights_kernel<<<(3 * DD * DD) / 256, 256, 0, stream>>>(Wq, Wk, Wv, wt);

  // 2) projections: 3 * 256 * 16 = 12288 waves
  proj_kernel<<<(12288 * 32) / 256, 256, 0, stream>>>(q, k, v, wt, bq, bk, bv,
                                                      Qp, Kp, Vt);

  // 3) attention: B*H*(S/16) = 4096 waves, 2 waves per block
  attn_kernel<<<4096 / 2, 64, 0, stream>>>(Qp, Kp, Vt, mask, out);
}
